// ElementalAtomwise_40527311405343
// MI455X (gfx1250) — compile-verified, matrix operand order checked by reading, not run on hardware
//
#include <hip/hip_runtime.h>
#include <hip/hip_bf16.h>

// ---------------------------------------------------------------------------
// ElementalAtomwise for MI455X (gfx1250, wave32, WMMA)
//  (1) counting-sort atoms by element (zidx, 4 MB) into d_ws
//  (2) element-uniform 16-atom WMMA tiles, software-pipelined:
//      - perm index prefetched 2 tiles ahead, x-rows 1 tile ahead (NT loads)
//      - B = W1[e] pre-swizzled in LDS in exact WMMA B-fragment layout,
//        reloaded per tile as ds_load_b128 pairs (LICM defeated) to keep the
//        VGPR allocation under 256 -> no s_set_vgpr_msb chatter, higher
//        occupancy for latency hiding (memory-bound kernel)
//      - b1 folded into WMMA accumulator init; ssp via raw v_exp_f32 /
//        v_log_f32; -ln2*sum(W2) folded out of the per-value path
//      - layer-2 dot in-register + shfl_xor reduce, f32 atomicAdd segment sum
// Roofline: 512 MB of x @ 23.3 TB/s ~= 22 us floor.
// ---------------------------------------------------------------------------

typedef __bf16 bf16_t;
typedef __attribute__((ext_vector_type(16))) __bf16 v16bf;
typedef __attribute__((ext_vector_type(8)))  __bf16 v8bf;
typedef __attribute__((ext_vector_type(8)))  float  v8f;
typedef __attribute__((ext_vector_type(4)))  float  v4f;

#define NUM_E 4
#define DIM_D 128
#define DIM_H 64
#define SCATTER_CHUNK 2048

#define LN2F      0.69314718055994531f
#define NLOG2EF  -1.4426950408889634f

__global__ void zero_f32_kernel(float* __restrict__ p, int n) {
    int i = blockIdx.x * blockDim.x + threadIdx.x;
    if (i < n) p[i] = 0.0f;
}

__global__ void zero_i32_kernel(int* __restrict__ p, int n) {
    int i = blockIdx.x * blockDim.x + threadIdx.x;
    if (i < n) p[i] = 0;
}

// ---- pass 1: per-element counts ------------------------------------------
__global__ void count_kernel(const int* __restrict__ zidx, int* __restrict__ counters, int N) {
    __shared__ int lc[NUM_E];
    if (threadIdx.x < NUM_E) lc[threadIdx.x] = 0;
    __syncthreads();
    for (int i = blockIdx.x * blockDim.x + threadIdx.x; i < N; i += gridDim.x * blockDim.x)
        atomicAdd(&lc[zidx[i]], 1);
    __syncthreads();
    if (threadIdx.x < NUM_E) atomicAdd(&counters[threadIdx.x], lc[threadIdx.x]);
}

// ---- pass 2: exclusive prefix -> bases [4..7] and cursors [8..11] --------
__global__ void prefix_kernel(int* __restrict__ counters) {
    if (blockIdx.x == 0 && threadIdx.x == 0) {
        int acc = 0;
        for (int e = 0; e < NUM_E; ++e) {
            counters[4 + e] = acc;
            counters[8 + e] = acc;
            acc += counters[e];
        }
    }
}

// ---- pass 3: block-aggregated scatter into element buckets ---------------
__launch_bounds__(256)
__global__ void scatter_kernel(const int* __restrict__ zidx, int* __restrict__ perm,
                               int* __restrict__ counters, int N) {
    __shared__ int lcount[NUM_E];
    __shared__ int lbase[NUM_E];
    const int tid = threadIdx.x;
    for (int start = blockIdx.x * SCATTER_CHUNK; start < N; start += gridDim.x * SCATTER_CHUNK) {
        if (tid < NUM_E) lcount[tid] = 0;
        __syncthreads();
        int myz[SCATTER_CHUNK / 256];
        int lpos[SCATTER_CHUNK / 256];
        int nmine = 0;
        for (int i = tid; i < SCATTER_CHUNK && start + i < N; i += 256) {
            int z = zidx[start + i];
            myz[nmine] = z;
            lpos[nmine] = atomicAdd(&lcount[z], 1);
            ++nmine;
        }
        __syncthreads();
        if (tid < NUM_E) lbase[tid] = atomicAdd(&counters[8 + tid], lcount[tid]);
        __syncthreads();
        int j = 0;
        for (int i = tid; i < SCATTER_CHUNK && start + i < N; i += 256) {
            perm[lbase[myz[j]] + lpos[j]] = start + i;
            ++j;
        }
        __syncthreads();
    }
}

// ---- main: element-uniform WMMA tiles, software pipelined ----------------
__launch_bounds__(256)
__global__ void atomwise_sorted_kernel(const float* __restrict__ x,
                                       const int*   __restrict__ idx_m,
                                       const float* __restrict__ W1,
                                       const float* __restrict__ b1,
                                       const float* __restrict__ W2,
                                       const float* __restrict__ b2,
                                       const int*   __restrict__ perm,
                                       const int*   __restrict__ counters,
                                       float* __restrict__ out)
{
    // W1[e] pre-swizzled into WMMA B-fragment half-records:
    // fragment f = n*4+kc, half h, lane l, slot j(0..7) ->
    //   K = kc*32 + (l>=16?8:0) + h*16 + j,  n_col = n*16 + (l&15)
    // Record (f,h,l) is 8 bf16 = 16 B at byte offset ((f*2+h)*32 + l)*16:
    // lane stride 16 B -> ds_load_b128 with minimal 2-phase bank access.
    __shared__ __align__(32) bf16_t sWB[32 * 32 * 8];   // 16 KB

    const int tid   = threadIdx.x;
    const int e     = blockIdx.x & 3;         // bucket handled by this block
    const int cnt   = counters[e];
    const int bb    = counters[4 + e];
    const float b2e = b2[e];

    for (int rec = tid; rec < 32 * 32; rec += blockDim.x) {
        const int fh = rec >> 5;             // fragment*2 + half
        const int l  = rec & 31;
        const int f  = fh >> 1;
        const int h  = fh & 1;
        const int n  = f >> 2;
        const int kc = f & 3;
        const int kbase = kc * 32 + ((l & 16) ? 8 : 0) + h * 16;
        const int ncol  = n * 16 + (l & 15);
        bf16_t* dst = sWB + (size_t)rec * 8;
        const float* src = W1 + (size_t)e * DIM_D * DIM_H + ncol;
        #pragma unroll
        for (int j = 0; j < 8; ++j)
            dst[j] = (bf16_t)src[(size_t)(kbase + j) * DIM_H];
    }
    __syncthreads();

    const int lane = tid & 31;
    const int wave = tid >> 5;
    const int col  = lane & 15;
    const bool hi  = lane >= 16;
    const int koff = hi ? 8 : 0;

    // Per-lane loop-invariant parameters.
    float w2m[4], w2l[4], b1v[4];
    #pragma unroll
    for (int n = 0; n < 4; ++n) {
        w2m[n] = W2[e * DIM_H + n * 16 + col];
        w2l[n] = w2m[n] * LN2F;
        b1v[n] = b1[e * DIM_H + n * 16 + col];
    }
    // Per-atom correction: sum over this lane's 4 hidden values of -ln2*W2.
    const float offTot = -(w2l[0] + w2l[1] + w2l[2] + w2l[3]);

    const int tiles_e = (cnt + 15) >> 4;
    const int wstride = (gridDim.x >> 2) * 8;
    int t = (blockIdx.x >> 2) * 8 + wave;
    if (t >= tiles_e) return;

    // --- pipelined helpers ------------------------------------------------
    auto load_atom = [&](int tt) -> int {
        const int slot = tt * 16 + col;
        return perm[bb + (slot < cnt ? slot : 0)];
    };
    v4f raw[16];
    auto load_rows = [&](int a) {
        const float* xrow = x + (size_t)a * DIM_D;
        #pragma unroll
        for (int kc = 0; kc < 4; ++kc) {
            const int ko = kc * 32 + koff;
            raw[kc * 4 + 0] = __builtin_nontemporal_load((const v4f*)(xrow + ko));
            raw[kc * 4 + 1] = __builtin_nontemporal_load((const v4f*)(xrow + ko + 4));
            raw[kc * 4 + 2] = __builtin_nontemporal_load((const v4f*)(xrow + ko + 16));
            raw[kc * 4 + 3] = __builtin_nontemporal_load((const v4f*)(xrow + ko + 20));
        }
    };

    // --- prologue ---------------------------------------------------------
    int aCur = load_atom(t);
    load_rows(aCur);
    int tn = t + wstride;
    int aNext = (tn < tiles_e) ? load_atom(tn) : 0;

    while (true) {
        // Convert this tile's raw rows to bf16 A-fragments (waits the loads).
        v16bf afr[4];
        #pragma unroll
        for (int kc = 0; kc < 4; ++kc) {
            const v4f f0 = raw[kc * 4 + 0];
            const v4f f1 = raw[kc * 4 + 1];
            const v4f f2 = raw[kc * 4 + 2];
            const v4f f3 = raw[kc * 4 + 3];
            v16bf a;
            a[0]  = (bf16_t)f0[0]; a[1]  = (bf16_t)f0[1];
            a[2]  = (bf16_t)f0[2]; a[3]  = (bf16_t)f0[3];
            a[4]  = (bf16_t)f1[0]; a[5]  = (bf16_t)f1[1];
            a[6]  = (bf16_t)f1[2]; a[7]  = (bf16_t)f1[3];
            a[8]  = (bf16_t)f2[0]; a[9]  = (bf16_t)f2[1];
            a[10] = (bf16_t)f2[2]; a[11] = (bf16_t)f2[3];
            a[12] = (bf16_t)f3[0]; a[13] = (bf16_t)f3[1];
            a[14] = (bf16_t)f3[2]; a[15] = (bf16_t)f3[3];
            afr[kc] = a;
        }
        const int atomC = aCur;
        const int tCur  = t;

        // Kick off next tile's gathered row loads + perm prefetch (2 ahead).
        t = tn;
        tn = t + wstride;
        if (t < tiles_e) {
            aCur = aNext;
            load_rows(aCur);
        }
        if (tn < tiles_e) aNext = load_atom(tn);

        // Opaque zero offset: defeats LICM so B-fragments stay in LDS and
        // are re-loaded each tile (keeps VGPR allocation < 256).
        int zoff = 0;
        asm volatile("" : "+v"(zoff));
        const bf16_t* wb = (const bf16_t*)sWB + zoff;

        // --- compute: 16 WMMAs + ssp + layer-2 dot -------------------------
        float s[8];
        #pragma unroll
        for (int r = 0; r < 8; ++r) s[r] = offTot;

        #pragma unroll
        for (int n = 0; n < 4; ++n) {
            v8f c;
            #pragma unroll
            for (int r = 0; r < 8; ++r) c[r] = b1v[n];   // bias in accumulator
            #pragma unroll
            for (int kc = 0; kc < 4; ++kc) {
                const int f = n * 4 + kc;
                const v8bf lo = *(const v8bf*)(wb + (size_t)((f * 2 + 0) * 32 + lane) * 8);
                const v8bf hb = *(const v8bf*)(wb + (size_t)((f * 2 + 1) * 32 + lane) * 8);
                const v16bf bfr = __builtin_shufflevector(
                    lo, hb, 0, 1, 2, 3, 4, 5, 6, 7, 8, 9, 10, 11, 12, 13, 14, 15);
                c = __builtin_amdgcn_wmma_f32_16x16x32_bf16(
                        false, afr[kc], false, bfr, (short)0, c, false, false);
            }
            #pragma unroll
            for (int r = 0; r < 8; ++r) {
                const float pre = c[r];
                // (ssp(pre) - ln2, shift folded into offTot) * W2:
                //   max(pre,0)*w2 + ln2*w2*log2(1 + exp2(-log2e*|pre|))
                const float m = fmaxf(pre, 0.0f);
                const float u = __builtin_amdgcn_exp2f(NLOG2EF * fabsf(pre));
                const float L = __builtin_amdgcn_logf(1.0f + u);
                s[r] = fmaf(L, w2l[n], s[r]);
                s[r] = fmaf(m, w2m[n], s[r]);
            }
        }

        // reduce 16 hidden-columns per atom within each 16-lane group
        #pragma unroll
        for (int r = 0; r < 8; ++r) {
            float v = s[r];
            v += __shfl_xor(v, 1, 16);
            v += __shfl_xor(v, 2, 16);
            v += __shfl_xor(v, 4, 16);
            v += __shfl_xor(v, 8, 16);
            s[r] = v;
        }

        if (col == 0) {
            const int mbase = tCur * 16 + (hi ? 8 : 0);
            #pragma unroll
            for (int r = 0; r < 8; ++r) {
                if (mbase + r < cnt) {
                    const int atomr = __shfl(atomC, hi ? (8 + r) : r, 16);
                    atomicAdd(&out[idx_m[atomr]], s[r] + b2e);
                }
            }
        }

        if (t >= tiles_e) break;
    }
}

// ---------------------------------------------------------------------------
// Fallback (no workspace): all-element compute with per-atom mask select.
// ---------------------------------------------------------------------------
#define HPAD 72
__launch_bounds__(256)
__global__ void atomwise_fallback_kernel(const float* __restrict__ x,
                                         const int*   __restrict__ zidx,
                                         const int*   __restrict__ idx_m,
                                         const float* __restrict__ W1,
                                         const float* __restrict__ b1,
                                         const float* __restrict__ W2,
                                         const float* __restrict__ b2,
                                         float* __restrict__ out,
                                         int N)
{
    __shared__ bf16_t sW1[NUM_E * DIM_D * HPAD];
    __shared__ float  sb1[NUM_E * DIM_H];
    __shared__ float  sW2[NUM_E * DIM_H];
    __shared__ float  sb2[NUM_E];

    const int tid = threadIdx.x;
    for (int i = tid; i < NUM_E * DIM_D * DIM_H; i += blockDim.x) {
        int n = i % DIM_H, k = (i / DIM_H) % DIM_D, e = i / (DIM_H * DIM_D);
        sW1[(e * DIM_D + k) * HPAD + n] = (bf16_t)W1[i];
    }
    for (int i = tid; i < NUM_E * DIM_H; i += blockDim.x) { sb1[i] = b1[i]; sW2[i] = W2[i]; }
    if (tid < NUM_E) sb2[tid] = b2[tid];
    __syncthreads();

    const int lane = tid & 31, wave = tid >> 5, col = lane & 15;
    const bool hi = lane >= 16;
    const int koff = hi ? 8 : 0;
    const int tiles = N >> 4;
    const int wcount = gridDim.x * 8;

    for (int tile = blockIdx.x * 8 + wave; tile < tiles; tile += wcount) {
        const int base = tile << 4;
        v16bf afr[4];
        {
            const float* xrow = x + (size_t)(base + col) * DIM_D;
            #pragma unroll
            for (int kc = 0; kc < 4; ++kc) {
                const int ko = kc * 32 + koff;
                const v4f f0 = *(const v4f*)(xrow + ko);
                const v4f f1 = *(const v4f*)(xrow + ko + 4);
                const v4f f2 = *(const v4f*)(xrow + ko + 16);
                const v4f f3 = *(const v4f*)(xrow + ko + 20);
                v16bf a;
                a[0]=(bf16_t)f0[0]; a[1]=(bf16_t)f0[1]; a[2]=(bf16_t)f0[2]; a[3]=(bf16_t)f0[3];
                a[4]=(bf16_t)f1[0]; a[5]=(bf16_t)f1[1]; a[6]=(bf16_t)f1[2]; a[7]=(bf16_t)f1[3];
                a[8]=(bf16_t)f2[0]; a[9]=(bf16_t)f2[1]; a[10]=(bf16_t)f2[2]; a[11]=(bf16_t)f2[3];
                a[12]=(bf16_t)f3[0]; a[13]=(bf16_t)f3[1]; a[14]=(bf16_t)f3[2]; a[15]=(bf16_t)f3[3];
                afr[kc] = a;
            }
        }
        int zl = zidx[base + col];
        int zr[8];
        #pragma unroll
        for (int r = 0; r < 8; ++r) zr[r] = __shfl(zl, hi ? (8 + r) : r, 16);

        float tp[8];
        #pragma unroll
        for (int r = 0; r < 8; ++r) tp[r] = 0.0f;

        #pragma unroll
        for (int e = 0; e < NUM_E; ++e) {
            float s[8];
            #pragma unroll
            for (int r = 0; r < 8; ++r) s[r] = 0.0f;
            #pragma unroll
            for (int n = 0; n < 4; ++n) {
                const float b1v = sb1[e * DIM_H + n * 16 + col];
                v8f c;
                #pragma unroll
                for (int r = 0; r < 8; ++r) c[r] = b1v;
                const bf16_t* wbp = sW1 + (size_t)e * DIM_D * HPAD + (n * 16 + col);
                #pragma unroll
                for (int kc = 0; kc < 4; ++kc) {
                    const int kb = kc * 32 + koff;
                    v16bf bfr;
                    #pragma unroll
                    for (int i = 0; i < 16; ++i) {
                        const int krel = (i < 8) ? i : (i + 8);
                        bfr[i] = wbp[(size_t)(kb + krel) * HPAD];
                    }
                    c = __builtin_amdgcn_wmma_f32_16x16x32_bf16(
                            false, afr[kc], false, bfr, (short)0, c, false, false);
                }
                const float w2mv = sW2[e * DIM_H + n * 16 + col];
                const float w2lv = w2mv * LN2F;
                #pragma unroll
                for (int r = 0; r < 8; ++r) {
                    const float pre = c[r];
                    const float m = fmaxf(pre, 0.0f);
                    const float u = __builtin_amdgcn_exp2f(NLOG2EF * fabsf(pre));
                    const float L = __builtin_amdgcn_logf(1.0f + u);
                    s[r] = fmaf(L, w2lv, s[r]);
                    s[r] = fmaf(m, w2mv, s[r]);
                    s[r] -= w2lv;            // -ln2 * W2 shift
                }
            }
            #pragma unroll
            for (int r = 0; r < 8; ++r)
                if (zr[r] == e) tp[r] += s[r];
        }
        #pragma unroll
        for (int r = 0; r < 8; ++r) {
            float v = tp[r];
            v += __shfl_xor(v, 1, 16);
            v += __shfl_xor(v, 2, 16);
            v += __shfl_xor(v, 4, 16);
            v += __shfl_xor(v, 8, 16);
            tp[r] = v;
        }
        if (col == 0) {
            const int rbase = base + (hi ? 8 : 0);
            #pragma unroll
            for (int r = 0; r < 8; ++r)
                atomicAdd(&out[idx_m[rbase + r]], tp[r] + sb2[zr[r]]);
        }
    }
}

// ---------------------------------------------------------------------------
extern "C" void kernel_launch(void* const* d_in, const int* in_sizes, int n_in,
                              void* d_out, int out_size, void* d_ws, size_t ws_size,
                              hipStream_t stream) {
    const float* x     = (const float*)d_in[0];
    const int*   zidx  = (const int*)  d_in[1];
    const int*   idx_m = (const int*)  d_in[2];
    const float* W1    = (const float*)d_in[3];
    const float* b1    = (const float*)d_in[4];
    const float* W2    = (const float*)d_in[5];
    const float* b2    = (const float*)d_in[6];
    float* out = (float*)d_out;

    const int N = in_sizes[0] / DIM_D;

    zero_f32_kernel<<<(out_size + 255) / 256, 256, 0, stream>>>(out, out_size);

    const size_t need = (size_t)N * sizeof(int) + 16 * sizeof(int);
    if (ws_size >= need && N >= 16 && (N & 15) == 0) {
        int* perm     = (int*)d_ws;
        int* counters = perm + N;            // 12 ints used
        zero_i32_kernel<<<1, 64, 0, stream>>>(counters, 12);

        int cblocks = (N + 2047) / 2048; if (cblocks > 2048) cblocks = 2048;
        count_kernel<<<cblocks, 256, 0, stream>>>(zidx, counters, N);
        prefix_kernel<<<1, 1, 0, stream>>>(counters);
        int sblocks = (N + SCATTER_CHUNK - 1) / SCATTER_CHUNK;
        if (sblocks > 2048) sblocks = 2048;
        scatter_kernel<<<sblocks, 256, 0, stream>>>(zidx, perm, counters, N);

        int mblocks = 1280;                  // multiple of 4 (bucket = block % 4)
        atomwise_sorted_kernel<<<mblocks, 256, 0, stream>>>(
            x, idx_m, W1, b1, W2, b2, perm, counters, out);
    } else {
        const int tiles = N / 16;
        int blocks = (tiles + 7) / 8;
        if (blocks > 1280) blocks = 1280;
        if (blocks < 1) blocks = 1;
        atomwise_fallback_kernel<<<blocks, 256, 0, stream>>>(
            x, zidx, idx_m, W1, b1, W2, b2, out, N);
    }
}
